// LSM_conv_25881472926532
// MI455X (gfx1250) — matrix operand order repaired and checked by reading, not compiled, not run on hardware
//
#include <hip/hip_runtime.h>
#include <hip/hip_bf16.h>

typedef __attribute__((ext_vector_type(16))) _Float16 v16h;
typedef __attribute__((ext_vector_type(8)))  _Float16 v8h;
typedef __attribute__((ext_vector_type(8)))  float    v8f;

#define HW 1156              // 34*34
#define MROWS (32 * HW)      // 36992 = B*H*W
#define MTILES (MROWS / 16)  // 2312
#define PADHW (36 * 36)

// ---------------------------------------------------------------------------
// Prep: swizzle w_rec [64,64,3,3] into WMMA B-fragment layout.
// frag element index = ((nt*18 + kk)*32 + lane)*16 + j
//   K(global) = tap*64 + (kk&1)*32 + q, q = (j<8 ? 0 : 16) + (lane>=16 ? 8 : 0) + (j&7)
//   N = nt*16 + (lane&15)
__global__ void prep_wrec_frag(const float* __restrict__ w_rec,
                               _Float16* __restrict__ frag) {
  int tid = blockIdx.x * blockDim.x + threadIdx.x;   // 36864 total
  int j    = tid & 15;
  int lane = (tid >> 4) & 31;
  int kk   = (tid >> 9) % 18;
  int nt   = tid / (512 * 18);
  int q  = ((j < 8) ? 0 : 16) + ((lane >= 16) ? 8 : 0) + (j & 7);
  int n  = nt * 16 + (lane & 15);
  int tap = kk >> 1;
  int ci  = (kk & 1) * 32 + q;
  int dy = tap / 3, dx = tap % 3;
  frag[tid] = (_Float16)w_rec[((n * 64 + ci) * 3 + dy) * 3 + dx];
}

// w_r1 [10, 192, 3, 3] -> B-fragments, N padded 10->16 with zeros. K = 1728.
__global__ void prep_wr1_frag(const float* __restrict__ w_r1,
                              _Float16* __restrict__ frag) {
  int tid = blockIdx.x * blockDim.x + threadIdx.x;   // 27648 total
  int j    = tid & 15;
  int lane = (tid >> 4) & 31;
  int kk   = tid >> 9;                               // 0..53
  int q  = ((j < 8) ? 0 : 16) + ((lane >= 16) ? 8 : 0) + (j & 7);
  int n  = lane & 15;
  int tap = kk / 6;
  int ci  = (kk % 6) * 32 + q;                       // 0..191
  int dy = tap / 3, dx = tap % 3;
  float v = (n < 10) ? w_r1[((n * 192 + ci) * 3 + dy) * 3 + dx] : 0.0f;
  frag[tid] = (_Float16)v;
}

// ---------------------------------------------------------------------------
// One LSM timestep: recurrent 64->64 conv via WMMA implicit GEMM
//   M = b*1156 + y*34 + x (36992 rows), N = co (64), K = (tap,ci) (576)
// fused with the 2->64 input conv and LIF state update.
__global__ __launch_bounds__(256) void lsm_step(
    const float* __restrict__ x, int t,
    const float* __restrict__ w_in, const float* __restrict__ b_in,
    const float* __restrict__ b_rec,
    const _Float16* __restrict__ wrec_frag,
    const _Float16* __restrict__ spk_prev,   // padded NHWC f16 [32][36][36][64]
    _Float16* __restrict__ spk_next,
    float* __restrict__ syn, float* __restrict__ mem,  // NHWC [m][64]
    float* __restrict__ spk_rec)             // NCHW [T][32][64][34][34]
{
  int wave = threadIdx.x >> 5;
  int lane = threadIdx.x & 31;
  int mt = blockIdx.x * 2 + (wave >> 2);     // 2 M-tiles per block
  int nt = wave & 3;                         // 4 N-tiles (co groups)
  int l  = lane & 15;
  int hi = (lane >= 16) ? 1 : 0;
  int co = nt * 16 + l;

  // A-fragment row owned by this lane
  int mrow = mt * 16 + l;
  int b  = mrow / HW;
  int r  = mrow - b * HW;
  int ay = r / 34;
  int ax = r - ay * 34;
  const _Float16* abase = spk_prev + ((size_t)(b * 36 + ay) * 36 + ax) * 64;
  int aoff = hi ? 8 : 0;

  v8f acc = {0.f, 0.f, 0.f, 0.f, 0.f, 0.f, 0.f, 0.f};
  const v16h* bf = (const v16h*)wrec_frag + (size_t)nt * 18 * 32 + lane;

#pragma unroll
  for (int kk = 0; kk < 18; ++kk) {
    int tap = kk >> 1;
    int dy = tap / 3, dx = tap % 3;
    int h = (kk & 1) * 32;
    const _Float16* p = abase + (size_t)(dy * 36 + dx) * 64 + h + aoff;
    union { v16h v; v8h u[2]; } av;
    av.u[0] = *(const v8h*)(p);
    av.u[1] = *(const v8h*)(p + 16);
    v16h bv = bf[(size_t)kk * 32];
    acc = __builtin_amdgcn_wmma_f32_16x16x32_f16(
        false, av.v, false, bv, (short)0, acc, false, false);
  }

  // ---- fused epilogue: input conv (18 MACs/elt) + LIF update ----
  float wi[18];
#pragma unroll
  for (int q = 0; q < 18; ++q) wi[q] = w_in[co * 18 + q];
  float bias = b_in[co] + b_rec[co];
  const float* xt = x + (size_t)t * 32 * 2 * HW;
  float* srec = spk_rec + (size_t)t * 32 * 64 * HW;

#pragma unroll
  for (int i = 0; i < 8; ++i) {
    int m  = mt * 16 + i + hi * 8;
    int bb = m / HW;
    int r2 = m - bb * HW;
    int yy = r2 / 34;
    int xx = r2 - yy * 34;

    float cur = bias;
#pragma unroll
    for (int ci = 0; ci < 2; ++ci) {
      const float* xb = xt + ((size_t)bb * 2 + ci) * HW;
#pragma unroll
      for (int dy = 0; dy < 3; ++dy) {
        int py = yy + dy - 1;
        if (py < 0 || py >= 34) continue;
#pragma unroll
        for (int dx = 0; dx < 3; ++dx) {
          int px = xx + dx - 1;
          if (px < 0 || px >= 34) continue;
          cur += xb[py * 34 + px] * wi[ci * 9 + dy * 3 + dx];
        }
      }
    }

    size_t sidx = (size_t)m * 64 + co;
    float sv = syn[sidx], mv = mem[sidx];
    float rst = (mv > 1.0f) ? 1.0f : 0.0f;
    float sn = 0.9f * sv + cur + acc[i];
    float mn = 0.9f * mv + sn - rst;
    float sp = (mn > 1.0f) ? 1.0f : 0.0f;
    syn[sidx] = sn;
    mem[sidx] = mn;
    srec[((size_t)bb * 64 + co) * HW + r2] = sp;
    spk_next[((size_t)(bb * 36 + yy + 1) * 36 + (xx + 1)) * 64 + co] = (_Float16)sp;
  }
}

// ---------------------------------------------------------------------------
// Pool 16 timesteps per partition into padded NHWC f16 [32][36][36][192]
__global__ void lsm_meanpack(const float* __restrict__ spk_rec,
                             _Float16* __restrict__ ro_in) {
  int tid = blockIdx.x * blockDim.x + threadIdx.x;   // 32*3*64*1156 = 7102464
  int x = tid % 34;
  int t1 = tid / 34;
  int y = t1 % 34;  t1 /= 34;
  int cc = t1 % 64; t1 /= 64;
  int p = t1 % 3;
  int b = t1 / 3;
  float s = 0.f;
#pragma unroll 4
  for (int j = 0; j < 16; ++j)
    s += spk_rec[(((size_t)(p * 16 + j) * 32 + b) * 64 + cc) * HW + y * 34 + x];
  ro_in[((size_t)(b * 36 + y + 1) * 36 + (x + 1)) * 192 + p * 64 + cc] =
      (_Float16)(s * 0.0625f);
}

// ---------------------------------------------------------------------------
// Readout conv 192->10 via WMMA (N padded to 16), K = 1728 (54 chunks)
__global__ __launch_bounds__(256) void lsm_readout_conv(
    const _Float16* __restrict__ ro_in, const _Float16* __restrict__ wr1_frag,
    const float* __restrict__ b_r1, float* __restrict__ y_conv) {
  int wave = threadIdx.x >> 5;
  int lane = threadIdx.x & 31;
  int mt = blockIdx.x * 8 + wave;
  int l = lane & 15;
  int hi = (lane >= 16) ? 1 : 0;
  int co = l;

  int mrow = mt * 16 + l;
  int b = mrow / HW;
  int r = mrow - b * HW;
  int ay = r / 34;
  int ax = r - ay * 34;
  const _Float16* abase = ro_in + ((size_t)(b * 36 + ay) * 36 + ax) * 192;
  int aoff = hi ? 8 : 0;

  v8f acc = {0.f, 0.f, 0.f, 0.f, 0.f, 0.f, 0.f, 0.f};
  const v16h* bf = (const v16h*)wr1_frag + lane;

#pragma unroll 6
  for (int kk = 0; kk < 54; ++kk) {
    int tap = kk / 6;
    int dy = tap / 3, dx = tap % 3;
    int h = (kk % 6) * 32;
    const _Float16* p = abase + (size_t)(dy * 36 + dx) * 192 + h + aoff;
    union { v16h v; v8h u[2]; } av;
    av.u[0] = *(const v8h*)(p);
    av.u[1] = *(const v8h*)(p + 16);
    v16h bv = bf[(size_t)kk * 32];
    acc = __builtin_amdgcn_wmma_f32_16x16x32_f16(
        false, av.v, false, bv, (short)0, acc, false, false);
  }

  if (co < 10) {
    float bias = b_r1[co];
#pragma unroll
    for (int i = 0; i < 8; ++i) {
      int m = mt * 16 + i + hi * 8;
      int bb = m / HW;
      int r2 = m - bb * HW;
      y_conv[((size_t)bb * 10 + co) * HW + r2] = acc[i] + bias;
    }
  }
}

// ---------------------------------------------------------------------------
// Final FC [B,11560] @ w_r2.T + b_r2, then softmax. One block per batch elem.
__global__ __launch_bounds__(256) void lsm_fc_softmax(
    const float* __restrict__ y_conv, const float* __restrict__ w_r2,
    const float* __restrict__ b_r2, float* __restrict__ out) {
  __shared__ float red[256];
  __shared__ float logit[10];
  int b = blockIdx.x;
  const float* yb = y_conv + (size_t)b * 11560;
  for (int o = 0; o < 10; ++o) {
    float s = 0.f;
    for (int k = threadIdx.x; k < 11560; k += 256)
      s += yb[k] * w_r2[(size_t)o * 11560 + k];
    red[threadIdx.x] = s;
    __syncthreads();
    for (int st = 128; st > 0; st >>= 1) {
      if ((int)threadIdx.x < st) red[threadIdx.x] += red[threadIdx.x + st];
      __syncthreads();
    }
    if (threadIdx.x == 0) logit[o] = red[0] + b_r2[o];
    __syncthreads();
  }
  if (threadIdx.x == 0) {
    float mx = logit[0];
    for (int o = 1; o < 10; ++o) mx = fmaxf(mx, logit[o]);
    float e[10], sum = 0.f;
    for (int o = 0; o < 10; ++o) { e[o] = __expf(logit[o] - mx); sum += e[o]; }
    for (int o = 0; o < 10; ++o) out[b * 10 + o] = e[o] / sum;
  }
}

// ---------------------------------------------------------------------------
extern "C" void kernel_launch(void* const* d_in, const int* in_sizes, int n_in,
                              void* d_out, int out_size, void* d_ws, size_t ws_size,
                              hipStream_t stream) {
  const float* x     = (const float*)d_in[0];  // [48,32,2,34,34]
  const float* w_in  = (const float*)d_in[1];  // [64,2,3,3]
  const float* b_in  = (const float*)d_in[2];  // [64]
  const float* w_rec = (const float*)d_in[3];  // [64,64,3,3]
  const float* b_rec = (const float*)d_in[4];  // [64]
  const float* w_r1  = (const float*)d_in[5];  // [10,192,3,3]
  const float* b_r1  = (const float*)d_in[6];  // [10]
  const float* w_r2  = (const float*)d_in[7];  // [10,11560]
  const float* b_r2  = (const float*)d_in[8];  // [10]

  float* out     = (float*)d_out;          // [32,10]
  float* spk_rec = out + 320;              // [48,32,64,34,34]

  // ---- workspace layout (256B aligned) ----
  char* ws = (char*)d_ws;
  size_t off = 0;
  auto take = [&](size_t bytes) { char* p = ws + off; off = (off + bytes + 255) & ~(size_t)255; return p; };
  const size_t padBytes  = (size_t)32 * PADHW * 64 * 2;   // 5,308,416
  const size_t stBytes   = (size_t)MROWS * 64 * 4;        // 9,469,952
  _Float16* spk_pad0  = (_Float16*)take(padBytes);
  _Float16* spk_pad1  = (_Float16*)take(padBytes);
  float*    syn       = (float*)take(stBytes);
  float*    mem       = (float*)take(stBytes);
  _Float16* wrec_frag = (_Float16*)take(4 * 18 * 32 * 16 * 2);
  _Float16* wr1_frag  = (_Float16*)take(54 * 32 * 16 * 2);
  _Float16* ro_in     = (_Float16*)take((size_t)32 * PADHW * 192 * 2);
  float*    y_conv    = (float*)take((size_t)32 * 10 * HW * 4);

  // ---- zero state + halo regions (stream-ordered, capture-safe) ----
  hipMemsetAsync(spk_pad0, 0, 2 * padBytes + 2 * stBytes, stream);  // pads + syn + mem (contiguous)
  hipMemsetAsync(ro_in, 0, (size_t)32 * PADHW * 192 * 2, stream);

  // ---- one-time weight swizzles ----
  prep_wrec_frag<<<144, 256, 0, stream>>>(w_rec, wrec_frag);
  prep_wr1_frag<<<108, 256, 0, stream>>>(w_r1, wr1_frag);

  // ---- 48 sequential LIF steps (WMMA recurrent conv) ----
  for (int t = 0; t < 48; ++t) {
    const _Float16* prev = (t & 1) ? spk_pad1 : spk_pad0;
    _Float16*       next = (t & 1) ? spk_pad0 : spk_pad1;
    lsm_step<<<MTILES / 2, 256, 0, stream>>>(x, t, w_in, b_in, b_rec,
                                             wrec_frag, prev, next,
                                             syn, mem, spk_rec);
  }

  // ---- temporal pooling -> padded NHWC f16 ----
  lsm_meanpack<<<(32 * 3 * 64 * HW) / 256, 256, 0, stream>>>(spk_rec, ro_in);

  // ---- readout conv (WMMA) ----
  lsm_readout_conv<<<MTILES / 8, 256, 0, stream>>>(ro_in, wr1_frag, b_r1, y_conv);

  // ---- FC + softmax ----
  lsm_fc_softmax<<<32, 256, 0, stream>>>(y_conv, w_r2, b_r2, out);
}
